// CausalSelfAttention_57140244905985
// MI455X (gfx1250) — matrix operand order repaired and checked by reading, not compiled, not run on hardware
//
#include <hip/hip_runtime.h>

typedef unsigned short ushort_t;
typedef unsigned int   uint_t;

typedef __attribute__((ext_vector_type(16))) __bf16 v16bf;
typedef __attribute__((ext_vector_type(8)))  float  v8f;
typedef __attribute__((ext_vector_type(4)))  unsigned int v4u;
typedef __attribute__((ext_vector_type(8)))  int v8i;
typedef __attribute__((ext_vector_type(4)))  int v4i;

union ABFrag {
    uint4 q[2];
    v16bf v;
};

__device__ __forceinline__ ushort_t f2bf(float f) {
    uint_t u = __float_as_uint(f);
    uint_t r = u + 0x7FFFu + ((u >> 16) & 1u);   // round-to-nearest-even
    return (ushort_t)(r >> 16);
}

#define NEMB 256
#define SEQT 8192
#define NROW 16384   // B*T
#define CHUNK 64     // keys per iteration
#define NITER (SEQT / CHUNK)

// ---------------- Kernel 1: x f32 -> bf16 ----------------
__global__ void cvt_x_kernel(const float* __restrict__ x, ushort_t* __restrict__ xb) {
    int i = blockIdx.x * blockDim.x + threadIdx.x;
    xb[i] = f2bf(x[i]);
}

// ------- Kernel 2: W f32 [k][n] -> Wt bf16 [mat][n][k] -------
__global__ void cvt_w_kernel(const float* __restrict__ Wq, const float* __restrict__ Wk,
                             const float* __restrict__ Wv, ushort_t* __restrict__ wt) {
    int i = blockIdx.x * blockDim.x + threadIdx.x;   // i < 3*256*256
    int m = i >> 16;
    int r = i & 65535;
    int n = r >> 8, k = r & 255;
    const float* W = (m == 0) ? Wq : (m == 1) ? Wk : Wv;
    wt[i] = f2bf(W[k * NEMB + n]);
}

// ------- Kernel 3: projections  y = x@W + b  (WMMA bf16) -------
__global__ __launch_bounds__(256) void proj_kernel(
    const ushort_t* __restrict__ xb, const ushort_t* __restrict__ wt,
    const float* __restrict__ bq, const float* __restrict__ bk, const float* __restrict__ bv,
    ushort_t* __restrict__ qb, ushort_t* __restrict__ kbuf, ushort_t* __restrict__ vt)
{
    const int mat  = blockIdx.y;
    const int t0   = blockIdx.x * 16;
    const int lane = threadIdx.x & 31;
    const int w    = threadIdx.x >> 5;
    const int lr   = lane & 15;
    const int h    = lane >> 4;
    const int n0   = w * 32;

    const ushort_t* wmat = wt + mat * (NEMB * NEMB);
    const float*    bias = (mat == 0) ? bq : (mat == 1) ? bk : bv;

    ABFrag a[8];
    const ushort_t* arow = xb + (t0 + lr) * NEMB;
#pragma unroll
    for (int kk = 0; kk < 8; ++kk) {
        int kb_ = kk * 32 + h * 8;
        a[kk].q[0] = *(const uint4*)(arow + kb_);
        a[kk].q[1] = *(const uint4*)(arow + kb_ + 16);
    }

    v8f c0 = {}; v8f c1 = {};
    const ushort_t* brow0 = wmat + (n0 + lr) * NEMB;
    const ushort_t* brow1 = wmat + (n0 + 16 + lr) * NEMB;
#pragma unroll
    for (int kk = 0; kk < 8; ++kk) {
        int kb_ = kk * 32 + h * 8;
        ABFrag b0, b1;
        b0.q[0] = *(const uint4*)(brow0 + kb_);
        b0.q[1] = *(const uint4*)(brow0 + kb_ + 16);
        b1.q[0] = *(const uint4*)(brow1 + kb_);
        b1.q[1] = *(const uint4*)(brow1 + kb_ + 16);
        c0 = __builtin_amdgcn_wmma_f32_16x16x32_bf16(false, a[kk].v, false, b0.v, (short)0, c0, false, false);
        c1 = __builtin_amdgcn_wmma_f32_16x16x32_bf16(false, a[kk].v, false, b1.v, (short)0, c1, false, false);
    }

    float bias0 = bias[n0 + lr];
    float bias1 = bias[n0 + 16 + lr];

    if (mat < 2) {
        ushort_t* out = (mat == 0) ? qb : kbuf;
#pragma unroll
        for (int r = 0; r < 8; ++r) {
            int row = t0 + r + h * 8;
            out[row * NEMB + n0 + lr]      = f2bf(c0[r] + bias0);
            out[row * NEMB + n0 + 16 + lr] = f2bf(c1[r] + bias1);
        }
    } else {
        int batch = t0 >> 13;
        ushort_t* vb = vt + batch * (NEMB * SEQT);
#pragma unroll
        for (int r = 0; r < 8; ++r) {
            int tloc = (t0 + r + h * 8) & (SEQT - 1);
            vb[(n0 + lr) * SEQT + tloc]      = f2bf(c0[r] + bias0);
            vb[(n0 + 16 + lr) * SEQT + tloc] = f2bf(c1[r] + bias1);
        }
    }
}

// ---- TDM: issue one 2-D tile DMA (global -> LDS), D# per ISA 8.3/8.4 ----
__device__ __forceinline__ void tdm_load_2d(unsigned lds_addr, const void* gaddr,
                                            unsigned tensor_d0, unsigned tensor_d1,
                                            unsigned tile_d0, unsigned tile_d1,
                                            unsigned stride0) {
    unsigned long long ga = (unsigned long long)(uintptr_t)gaddr;
    v4u g0 = { 1u,                                   // count=1, user descriptor
               lds_addr,                             // LDS byte address
               (unsigned)ga,                         // global_addr[31:0]
               (unsigned)((ga >> 32) & 0x1FFFFFFu) | (2u << 30) };  // addr[56:32] | type=2
    v8i g1 = { (int)0x00010000u,                                    // data_size=1 (2 bytes)
               (int)((tensor_d0 & 0xFFFFu) << 16),                  // tensor_dim0[15:0]
               (int)((tensor_d0 >> 16) | ((tensor_d1 & 0xFFFFu) << 16)),
               (int)((tensor_d1 >> 16) | (tile_d0 << 16)),          // tile_dim0
               (int)(tile_d1 & 0xFFFFu),                            // tile_dim1 (tile_dim2=0)
               (int)stride0,                                        // tensor_dim0_stride[31:0]
               0, 0 };
    v4i z4 = { 0, 0, 0, 0 };
    v8i z8 = { 0, 0, 0, 0, 0, 0, 0, 0 };
    __builtin_amdgcn_tensor_load_to_lds(g0, g1, z4, z4, z8, 0);
}

// ------- Kernel 4: fused flash attention, TDM double-buffered K/V in LDS -------
// grid (64, 2); block 256 (8 waves). Wave handles 16 query rows; block shares
// 64-key K/V chunks staged by the Tensor Data Mover.
__global__ __launch_bounds__(256) void attn_kernel(
    const ushort_t* __restrict__ qb, const ushort_t* __restrict__ kbuf,
    const ushort_t* __restrict__ vt, float* __restrict__ out)
{
    extern __shared__ __attribute__((aligned(16))) ushort_t smem[];
    // layout (ushort elems): K[2] @ 0/16384, V[2] @ 32768/49152, pstage @ 65536
    const unsigned KOFF[2] = { 0u, 16384u };
    const unsigned VOFF[2] = { 32768u, 49152u };
    const unsigned POFF    = 65536u;

    const int lane  = threadIdx.x & 31;
    const int w     = threadIdx.x >> 5;
    const int lr    = lane & 15;
    const int h     = lane >> 4;
    const int batch = blockIdx.y;
    const int t0    = blockIdx.x * 128 + w * 16;

    const ushort_t* Q = qb   + (batch * SEQT + t0) * NEMB;
    const ushort_t* K = kbuf + (size_t)batch * SEQT * NEMB;
    const ushort_t* V = vt   + (size_t)batch * NEMB * SEQT;

    // Q A-fragments held in registers (16 rows x 256 d)
    ABFrag qa[8];
    const ushort_t* qrow = Q + lr * NEMB;
#pragma unroll
    for (int kk = 0; kk < 8; ++kk) {
        int kb_ = kk * 32 + h * 8;
        qa[kk].q[0] = *(const uint4*)(qrow + kb_);
        qa[kk].q[1] = *(const uint4*)(qrow + kb_ + 16);
    }

    v8f o[16];
    {
        v8f z = {};
#pragma unroll
        for (int j = 0; j < 16; ++j) o[j] = z;
    }
    float m[8], l[8];
#pragma unroll
    for (int r = 0; r < 8; ++r) { m[r] = -__builtin_inff(); l[r] = 0.0f; }

    // prologue: wave 0 DMAs chunk 0 (K tile 64x256, V tile 256x64)
    if (w == 0) {
        tdm_load_2d((unsigned)(uintptr_t)&smem[KOFF[0]], K,
                    NEMB, SEQT, NEMB, CHUNK, NEMB);
        tdm_load_2d((unsigned)(uintptr_t)&smem[VOFF[0]], V,
                    SEQT, NEMB, CHUNK, NEMB, SEQT);
    }

    for (int it = 0; it < NITER; ++it) {
        const int buf = it & 1;
        if (w == 0) {
            if (it + 1 < NITER) {
                const int nb = (it + 1) & 1;
                const int sn = (it + 1) * CHUNK;
                tdm_load_2d((unsigned)(uintptr_t)&smem[KOFF[nb]], K + (size_t)sn * NEMB,
                            NEMB, SEQT, NEMB, CHUNK, NEMB);
                tdm_load_2d((unsigned)(uintptr_t)&smem[VOFF[nb]], V + sn,
                            SEQT, NEMB, CHUNK, NEMB, SEQT);
                __builtin_amdgcn_s_wait_tensorcnt(2);   // current chunk landed
            } else {
                __builtin_amdgcn_s_wait_tensorcnt(0);
            }
        }
        __syncthreads();   // chunk `it` visible to all waves

        const ushort_t* kc = &smem[KOFF[buf]];   // [64][256]
        const ushort_t* vc = &smem[VOFF[buf]];   // [256][64]

        // ---- S = Q K^T : 4 x 16-key tiles ----
        v8f c[4];
        {
            v8f z = {};
#pragma unroll
            for (int t = 0; t < 4; ++t) c[t] = z;
        }
#pragma unroll
        for (int kk = 0; kk < 8; ++kk) {
            int kb_ = kk * 32 + h * 8;
#pragma unroll
            for (int t = 0; t < 4; ++t) {
                const ushort_t* krow = kc + (t * 16 + lr) * NEMB + kb_;
                ABFrag b0;
                b0.q[0] = *(const uint4*)(krow);
                b0.q[1] = *(const uint4*)(krow + 16);
                c[t] = __builtin_amdgcn_wmma_f32_16x16x32_bf16(false, qa[kk].v, false, b0.v, (short)0, c[t], false, false);
            }
        }

        // ---- online softmax over 64 keys ----
        ushort_t* prow_w = &smem[POFF + w * (16 * CHUNK)];
#pragma unroll
        for (int r = 0; r < 8; ++r) {
            float mx = fmaxf(fmaxf(c[0][r], c[1][r]), fmaxf(c[2][r], c[3][r]));
            mx = fmaxf(mx, __shfl_xor(mx, 1));
            mx = fmaxf(mx, __shfl_xor(mx, 2));
            mx = fmaxf(mx, __shfl_xor(mx, 4));
            mx = fmaxf(mx, __shfl_xor(mx, 8));
            float mnew  = fmaxf(m[r], mx);
            float scale = __expf(m[r] - mnew);
            m[r] = mnew;
            float rs = 0.0f;
            ushort_t* pr = prow_w + (r + h * 8) * CHUNK + lr;
#pragma unroll
            for (int t = 0; t < 4; ++t) {
                float p = __expf(c[t][r] - mnew);
                pr[t * 16] = f2bf(p);               // C-layout -> LDS row-major
                rs += p;
            }
            rs += __shfl_xor(rs, 1);
            rs += __shfl_xor(rs, 2);
            rs += __shfl_xor(rs, 4);
            rs += __shfl_xor(rs, 8);
            l[r] = l[r] * scale + rs;
#pragma unroll
            for (int j = 0; j < 16; ++j) o[j][r] *= scale;
        }

        // ---- reload P as two bf16 A-fragments (16x32 each) ----
        ABFrag pa[2];
#pragma unroll
        for (int cch = 0; cch < 2; ++cch) {
            const ushort_t* pld = prow_w + lr * CHUNK + cch * 32 + h * 8;
            pa[cch].q[0] = *(const uint4*)(pld);
            pa[cch].q[1] = *(const uint4*)(pld + 16);
        }

        // ---- O += P V ----
#pragma unroll
        for (int j = 0; j < 16; ++j) {
            const ushort_t* vrow = vc + (j * 16 + lr) * CHUNK + h * 8;
#pragma unroll
            for (int cch = 0; cch < 2; ++cch) {
                ABFrag bvf;
                bvf.q[0] = *(const uint4*)(vrow + cch * 32);
                bvf.q[1] = *(const uint4*)(vrow + cch * 32 + 16);
                o[j] = __builtin_amdgcn_wmma_f32_16x16x32_bf16(false, pa[cch].v, false, bvf.v, (short)0, o[j], false, false);
            }
        }
        __syncthreads();   // done reading buf before it is re-DMA'd
    }

    // ---- epilogue: O / l, f32 store ----
#pragma unroll
    for (int r = 0; r < 8; ++r) {
        float inv = 1.0f / l[r];
        int row = t0 + r + h * 8;
        float* orow = out + ((size_t)batch * SEQT + row) * NEMB;
#pragma unroll
        for (int j = 0; j < 16; ++j) orow[j * 16 + lr] = o[j][r] * inv;
    }
}

extern "C" void kernel_launch(void* const* d_in, const int* in_sizes, int n_in,
                              void* d_out, int out_size, void* d_ws, size_t ws_size,
                              hipStream_t stream) {
    const float* x  = (const float*)d_in[0];
    const float* Wq = (const float*)d_in[1];
    const float* bq = (const float*)d_in[2];
    const float* Wk = (const float*)d_in[3];
    const float* bk = (const float*)d_in[4];
    const float* Wv = (const float*)d_in[5];
    const float* bv = (const float*)d_in[6];
    float* out = (float*)d_out;

    ushort_t* xb = (ushort_t*)d_ws;
    ushort_t* wt = xb + (size_t)NROW * NEMB;
    ushort_t* qb = wt + (size_t)3 * NEMB * NEMB;
    ushort_t* kb = qb + (size_t)NROW * NEMB;
    ushort_t* vt = kb + (size_t)NROW * NEMB;

    const int attn_lds = (65536 + 8 * 16 * CHUNK) * (int)sizeof(ushort_t);  // 147456 B
    (void)hipFuncSetAttribute((const void*)attn_kernel,
                              hipFuncAttributeMaxDynamicSharedMemorySize, attn_lds);

    cvt_x_kernel<<<(NROW * NEMB) / 256, 256, 0, stream>>>(x, xb);
    cvt_w_kernel<<<(3 * NEMB * NEMB) / 256, 256, 0, stream>>>(Wq, Wk, Wv, wt);
    proj_kernel<<<dim3(NROW / 16, 3), 256, 0, stream>>>(xb, wt, bq, bk, bv, qb, kb, vt);
    attn_kernel<<<dim3(SEQT / 128, 2), 256, attn_lds, stream>>>(qb, kb, vt, out);
}
